// Representation_8426725835214
// MI455X (gfx1250) — matrix operand order repaired and checked by reading, not compiled, not run on hardware
//
#include <hip/hip_runtime.h>
#include <math.h>

#define LV 24
#define NB 512
#define NCQ 13
#define BALL (LV*NB)      // 12288
#define IND 256
#define H1 128            // HIDDEN
#define H2 256            // HID
#define G1 (4*H1)         // 512 gate width
#define D2DIM 635
#define D2PAD 640

typedef __attribute__((ext_vector_type(16))) __bf16 v16bf;
typedef __attribute__((ext_vector_type(8)))  float  v8f;

union BF16Frag { v16bf v; unsigned u[8]; };

__device__ __forceinline__ unsigned bf16r(float x){
    unsigned u = __float_as_uint(x);
    return (u + 0x7FFFu + ((u >> 16) & 1u)) >> 16;   // RNE f32 -> bf16
}
__device__ __forceinline__ unsigned pk2(float a, float b){
    return bf16r(a) | (bf16r(b) << 16);
}
__device__ __forceinline__ float sigf(float x){ return 1.0f / (1.0f + expf(-x)); }

__device__ __forceinline__ v8f wmma_bf16(const BF16Frag& a, const BF16Frag& b, v8f c){
    return __builtin_amdgcn_wmma_f32_16x16x32_bf16(false, a.v, false, b.v,
                                                   (short)0, c, false, false);
}

// ---------------- weight packer: f32 [Nout,K] -> bf16-pair u32 [Nout, Kpad/2] ----
__global__ void k_pack_w(const float* __restrict__ W, unsigned* __restrict__ dst,
                         int Nout, int K, int Kpad, int total){
    int idx = blockIdx.x * blockDim.x + threadIdx.x;
    if (idx >= total) return;
    int KP = Kpad >> 1;
    int n = idx / KP, kp = idx - n * KP, k = kp * 2;
    float a = (k     < K) ? W[(size_t)n * K + k]     : 0.f;
    float b = (k + 1 < K) ? W[(size_t)n * K + k + 1] : 0.f;
    dst[idx] = pk2(a, b);
}

// ---------------- generic WMMA GEMM: Y = act(A[M,K] @ Wpk^T + bias) -------------
// block 256 = 8 waves; tile 64(M) x 128(N); wave w -> n-tile w, m-tiles 0..3
__global__ __launch_bounds__(256) void k_gemm_f32A(
    const float* __restrict__ A, int lda, int M, int K, int Kpad,
    const unsigned* __restrict__ Wpk, const float* __restrict__ bias,
    const float* __restrict__ rowmask, float* __restrict__ Y, int ldy, int relu)
{
    __shared__ unsigned As[64 * 16];
    const int tid = threadIdx.x;
    const int wave = tid >> 5, lane = tid & 31;
    const int lm = lane & 15, hi = lane >> 4;
    const int m0 = blockIdx.x * 64;
    const int n0 = blockIdx.y * 128 + wave * 16;
    const int KP = Kpad >> 1;
    const int n = n0 + lm;

    v8f acc[4];
    #pragma unroll
    for (int i = 0; i < 4; i++) {
        #pragma unroll
        for (int j = 0; j < 8; j++) acc[i][j] = 0.f;
    }

    for (int k0 = 0; k0 < Kpad; k0 += 32) {
        #pragma unroll
        for (int i = 0; i < 4; i++) {
            int lin = tid + 256 * i;
            int m = lin >> 4, kp = lin & 15;
            int k = k0 + 2 * kp;
            int gm = m0 + m;
            float a0 = 0.f, a1 = 0.f;
            if (gm < M) {
                if (k     < K) a0 = A[(size_t)gm * lda + k];
                if (k + 1 < K) a1 = A[(size_t)gm * lda + k + 1];
            }
            As[lin] = pk2(a0, a1);
        }
        __syncthreads();
        BF16Frag bf;
        #pragma unroll
        for (int v = 0; v < 8; v++)
            bf.u[v] = Wpk[(size_t)n * KP + (k0 >> 1) + ((v < 4) ? v : v + 4) + hi * 4];
        #pragma unroll
        for (int mt = 0; mt < 4; mt++) {
            BF16Frag af;
            #pragma unroll
            for (int v = 0; v < 8; v++)
                af.u[v] = As[(mt * 16 + lm) * 16 + ((v < 4) ? v : v + 4) + hi * 4];
            acc[mt] = wmma_bf16(af, bf, acc[mt]);
        }
        __syncthreads();
    }
    float bb = bias ? bias[n] : 0.f;
    #pragma unroll
    for (int mt = 0; mt < 4; mt++) {
        #pragma unroll
        for (int r = 0; r < 8; r++) {
            int m = m0 + mt * 16 + r + hi * 8;
            float val = acc[mt][r] + bb;
            if (relu) val = fmaxf(val, 0.f);
            if (rowmask) val *= rowmask[m];
            Y[(size_t)m * ldy + n] = val;
        }
    }
}

// ---------------- packed-A WMMA GEMM (A pre-packed bf16 pairs) ------------------
__global__ __launch_bounds__(256) void k_gemm_pkA(
    const unsigned* __restrict__ Apk, int KPa, int Kpad,
    const unsigned* __restrict__ Wpk, const float* __restrict__ bias,
    float* __restrict__ Y, int ldy)
{
    const int tid = threadIdx.x;
    const int wave = tid >> 5, lane = tid & 31;
    const int lm = lane & 15, hi = lane >> 4;
    const int m0 = blockIdx.x * 64;
    const int n0 = blockIdx.y * 128 + wave * 16;
    const int KP = Kpad >> 1;
    const int n = n0 + lm;

    v8f acc[4];
    #pragma unroll
    for (int i = 0; i < 4; i++) {
        #pragma unroll
        for (int j = 0; j < 8; j++) acc[i][j] = 0.f;
    }
    for (int k0 = 0; k0 < Kpad; k0 += 32) {
        BF16Frag bf;
        #pragma unroll
        for (int v = 0; v < 8; v++)
            bf.u[v] = Wpk[(size_t)n * KP + (k0 >> 1) + ((v < 4) ? v : v + 4) + hi * 4];
        #pragma unroll
        for (int mt = 0; mt < 4; mt++) {
            BF16Frag af;
            #pragma unroll
            for (int v = 0; v < 8; v++)
                af.u[v] = Apk[(size_t)(m0 + mt * 16 + lm) * KPa + (k0 >> 1) +
                              ((v < 4) ? v : v + 4) + hi * 4];
            acc[mt] = wmma_bf16(af, bf, acc[mt]);
        }
    }
    float bb = bias ? bias[n] : 0.f;
    #pragma unroll
    for (int mt = 0; mt < 4; mt++) {
        #pragma unroll
        for (int r = 0; r < 8; r++) {
            int m = m0 + mt * 16 + r + hi * 8;
            Y[(size_t)m * ldy + n] = acc[mt][r] + bb;
        }
    }
}

// ---------------- fused condition LSTM (13 steps, K=256 x-part + K=128 h-part) --
// block 512 = 16 waves; 64 batch rows/block; wave w: row-half (w&1), 64-col block (w>>1)
__global__ __launch_bounds__(512) void k_lstm1(
    const float* __restrict__ xseq,       // [BALL, NCQ, IND]
    const unsigned* __restrict__ Wih_pk,  // [512][128]
    const unsigned* __restrict__ Whh_pk,  // [512][64]
    const float* __restrict__ bih, const float* __restrict__ bhh,
    float* __restrict__ lo)               // [BALL, H1]
{
    extern __shared__ char smem[];
    float*    Gs = (float*)smem;                                  // 64*512 f32
    float*    Cs = (float*)(smem + 64 * 512 * 4);                 // 64*128 f32
    unsigned* Hs = (unsigned*)(smem + (64 * 512 + 64 * 128) * 4); // 64*64 u32 (packed h)
    unsigned* As = Hs + 64 * 64;                                  // 64*16 u32

    const int tid = threadIdx.x;
    const int wave = tid >> 5, lane = tid & 31;
    const int lm = lane & 15, hi = lane >> 4;
    const int mh = wave & 1, nb = wave >> 1;
    const int b0 = blockIdx.x * 64;

    for (int i = tid; i < 64 * 64; i += 512) Hs[i] = 0u;
    for (int i = tid; i < 64 * 128; i += 512) Cs[i] = 0.f;
    __syncthreads();

    for (int t = 0; t < NCQ; t++) {
        v8f acc[2][4];
        #pragma unroll
        for (int a = 0; a < 2; a++)
            #pragma unroll
            for (int b = 0; b < 4; b++) {
                #pragma unroll
                for (int j = 0; j < 8; j++) acc[a][b][j] = 0.f;
            }

        // x contribution: K = 256 in 8 chunks of 32, staged+packed through LDS
        for (int kc = 0; kc < 8; kc++) {
            int k0 = kc * 32;
            #pragma unroll
            for (int i = 0; i < 2; i++) {
                int lin = tid + 512 * i;
                int m = lin >> 4, kp = lin & 15;
                const float* src = xseq + ((size_t)(b0 + m) * NCQ + t) * IND + k0 + 2 * kp;
                As[lin] = pk2(src[0], src[1]);
            }
            __syncthreads();
            BF16Frag af[2];
            #pragma unroll
            for (int mt = 0; mt < 2; mt++) {
                int row = mh * 32 + mt * 16 + lm;
                #pragma unroll
                for (int v = 0; v < 8; v++)
                    af[mt].u[v] = As[row * 16 + ((v < 4) ? v : v + 4) + hi * 4];
            }
            #pragma unroll
            for (int ns = 0; ns < 4; ns++) {
                int n = nb * 64 + ns * 16 + lm;
                BF16Frag bf;
                #pragma unroll
                for (int v = 0; v < 8; v++)
                    bf.u[v] = Wih_pk[(size_t)n * 128 + (k0 >> 1) + ((v < 4) ? v : v + 4) + hi * 4];
                acc[0][ns] = wmma_bf16(af[0], bf, acc[0][ns]);
                acc[1][ns] = wmma_bf16(af[1], bf, acc[1][ns]);
            }
            __syncthreads();
        }
        // h contribution: K = 128 from LDS-resident packed h
        for (int kc = 0; kc < 4; kc++) {
            BF16Frag af[2];
            #pragma unroll
            for (int mt = 0; mt < 2; mt++) {
                int row = mh * 32 + mt * 16 + lm;
                #pragma unroll
                for (int v = 0; v < 8; v++)
                    af[mt].u[v] = Hs[row * 64 + kc * 16 + ((v < 4) ? v : v + 4) + hi * 4];
            }
            #pragma unroll
            for (int ns = 0; ns < 4; ns++) {
                int n = nb * 64 + ns * 16 + lm;
                BF16Frag bf;
                #pragma unroll
                for (int v = 0; v < 8; v++)
                    bf.u[v] = Whh_pk[(size_t)n * 64 + kc * 16 + ((v < 4) ? v : v + 4) + hi * 4];
                acc[0][ns] = wmma_bf16(af[0], bf, acc[0][ns]);
                acc[1][ns] = wmma_bf16(af[1], bf, acc[1][ns]);
            }
        }
        // gates -> LDS (+ biases)
        #pragma unroll
        for (int mt = 0; mt < 2; mt++)
            #pragma unroll
            for (int ns = 0; ns < 4; ns++) {
                int n = nb * 64 + ns * 16 + lm;
                float bb = bih[n] + bhh[n];
                #pragma unroll
                for (int r = 0; r < 8; r++) {
                    int row = mh * 32 + mt * 16 + r + hi * 8;
                    Gs[row * G1 + n] = acc[mt][ns][r] + bb;
                }
            }
        __syncthreads();
        // cell update (f32), h repacked to WMMA A-fragment bf16 layout
        #pragma unroll
        for (int p = 0; p < 8; p++) {
            int lin = tid + 512 * p;           // 0..4095 pairs
            int m = lin >> 6, jp = lin & 63;
            int j = jp * 2;
            float i0 = Gs[m * G1 + j],       i1 = Gs[m * G1 + j + 1];
            float f0 = Gs[m * G1 + 128 + j], f1 = Gs[m * G1 + 128 + j + 1];
            float g0 = Gs[m * G1 + 256 + j], g1 = Gs[m * G1 + 256 + j + 1];
            float o0 = Gs[m * G1 + 384 + j], o1 = Gs[m * G1 + 384 + j + 1];
            float c0 = sigf(f0) * Cs[m * H1 + j]     + sigf(i0) * tanhf(g0);
            float c1 = sigf(f1) * Cs[m * H1 + j + 1] + sigf(i1) * tanhf(g1);
            Cs[m * H1 + j] = c0; Cs[m * H1 + j + 1] = c1;
            float h0 = sigf(o0) * tanhf(c0);
            float h1 = sigf(o1) * tanhf(c1);
            Hs[m * 64 + jp] = pk2(h0, h1);
            if (t == NCQ - 1) {
                lo[(size_t)(b0 + m) * H1 + j]     = h0;
                lo[(size_t)(b0 + m) * H1 + j + 1] = h1;
            }
        }
        __syncthreads();
    }
}

// ---------------- tree-LSTM one level: gather children, h@Whh WMMA, cell --------
__global__ __launch_bounds__(512) void k_tree(
    const float* __restrict__ xw2lev,    // [NB][G1] for this level (b2ih already added)
    const int*   __restrict__ maplev,    // [NB][2] (ignored if first)
    const unsigned* __restrict__ Whh_pk, // [512][64]
    const float* __restrict__ bhh,
    const float* __restrict__ hprev, const float* __restrict__ cprev,
    float* __restrict__ hout, float* __restrict__ cout, int first)
{
    extern __shared__ char smem[];
    float*    Gs  = (float*)smem;                                  // 64*512
    float*    Cin = (float*)(smem + 64 * 512 * 4);                 // 64*128
    unsigned* Hin = (unsigned*)(smem + (64 * 512 + 64 * 128) * 4); // 64*64

    const int tid = threadIdx.x;
    const int wave = tid >> 5, lane = tid & 31;
    const int lm = lane & 15, hi = lane >> 4;
    const int mh = wave & 1, nb = wave >> 1;
    const int b0 = blockIdx.x * 64;

    // gather children (averaged), pack h into WMMA A layout
    #pragma unroll
    for (int p = 0; p < 8; p++) {
        int lin = tid + 512 * p;
        int m = lin >> 6, jp = lin & 63;
        int j = jp * 2;
        float h0 = 0.f, h1 = 0.f, c0 = 0.f, c1 = 0.f;
        if (!first) {
            int gn = b0 + m;
            int ia = maplev[gn * 2], ib = maplev[gn * 2 + 1];
            if (ia > 0) {
                h0 += hprev[(size_t)(ia - 1) * H1 + j]; h1 += hprev[(size_t)(ia - 1) * H1 + j + 1];
                c0 += cprev[(size_t)(ia - 1) * H1 + j]; c1 += cprev[(size_t)(ia - 1) * H1 + j + 1];
            }
            if (ib > 0) {
                h0 += hprev[(size_t)(ib - 1) * H1 + j]; h1 += hprev[(size_t)(ib - 1) * H1 + j + 1];
                c0 += cprev[(size_t)(ib - 1) * H1 + j]; c1 += cprev[(size_t)(ib - 1) * H1 + j + 1];
            }
            h0 *= 0.5f; h1 *= 0.5f; c0 *= 0.5f; c1 *= 0.5f;
        }
        Hin[m * 64 + jp] = pk2(h0, h1);
        Cin[m * H1 + j] = c0; Cin[m * H1 + j + 1] = c1;
    }
    __syncthreads();

    v8f acc[2][4];
    #pragma unroll
    for (int a = 0; a < 2; a++)
        #pragma unroll
        for (int b = 0; b < 4; b++) {
            #pragma unroll
            for (int j = 0; j < 8; j++) acc[a][b][j] = 0.f;
        }
    for (int kc = 0; kc < 4; kc++) {
        BF16Frag af[2];
        #pragma unroll
        for (int mt = 0; mt < 2; mt++) {
            int row = mh * 32 + mt * 16 + lm;
            #pragma unroll
            for (int v = 0; v < 8; v++)
                af[mt].u[v] = Hin[row * 64 + kc * 16 + ((v < 4) ? v : v + 4) + hi * 4];
        }
        #pragma unroll
        for (int ns = 0; ns < 4; ns++) {
            int n = nb * 64 + ns * 16 + lm;
            BF16Frag bf;
            #pragma unroll
            for (int v = 0; v < 8; v++)
                bf.u[v] = Whh_pk[(size_t)n * 64 + kc * 16 + ((v < 4) ? v : v + 4) + hi * 4];
            acc[0][ns] = wmma_bf16(af[0], bf, acc[0][ns]);
            acc[1][ns] = wmma_bf16(af[1], bf, acc[1][ns]);
        }
    }
    #pragma unroll
    for (int mt = 0; mt < 2; mt++)
        #pragma unroll
        for (int ns = 0; ns < 4; ns++) {
            int n = nb * 64 + ns * 16 + lm;
            float bb = bhh[n];
            #pragma unroll
            for (int r = 0; r < 8; r++) {
                int row = mh * 32 + mt * 16 + r + hi * 8;
                Gs[row * G1 + n] = acc[mt][ns][r] + xw2lev[(size_t)(b0 + row) * G1 + n] + bb;
            }
        }
    __syncthreads();
    #pragma unroll
    for (int p = 0; p < 16; p++) {
        int lin = tid + 512 * p;               // 0..8191
        int m = lin >> 7, j = lin & 127;
        float ig = Gs[m * G1 + j],       fg = Gs[m * G1 + 128 + j];
        float gg = Gs[m * G1 + 256 + j], og = Gs[m * G1 + 384 + j];
        float c = sigf(fg) * Cin[m * H1 + j] + sigf(ig) * tanhf(gg);
        float h = sigf(og) * tanhf(c);
        hout[(size_t)(b0 + m) * H1 + j] = h;
        cout[(size_t)(b0 + m) * H1 + j] = c;
    }
}

// ---------------- misc elementwise / reduction kernels --------------------------
__global__ void k_avg(const float* __restrict__ a, const float* __restrict__ b,
                      float* __restrict__ y, int n){
    int i = blockIdx.x * blockDim.x + threadIdx.x;
    if (i < n) y[i] = 0.5f * (a[i] + b[i]);
}

__global__ void k_bnstats(const float* __restrict__ X, int M, int F,
                          float* __restrict__ mean, float* __restrict__ rstd){
    int f = blockIdx.x;
    int tid = threadIdx.x;
    float s = 0.f, s2 = 0.f;
    for (int m = tid; m < M; m += blockDim.x) {
        float v = X[(size_t)m * F + f];
        s += v; s2 += v * v;
    }
    __shared__ float rs[256], rq[256];
    rs[tid] = s; rq[tid] = s2;
    __syncthreads();
    for (int off = 128; off > 0; off >>= 1) {
        if (tid < off) { rs[tid] += rs[tid + off]; rq[tid] += rq[tid + off]; }
        __syncthreads();
    }
    if (tid == 0) {
        float mu = rs[0] / (float)M;
        float var = rq[0] / (float)M - mu * mu;
        mean[f] = mu;
        rstd[f] = rsqrtf(var + 1e-5f);
    }
}

__global__ void k_bnapply(const float* __restrict__ X, float* __restrict__ Y,
                          const float* __restrict__ mean, const float* __restrict__ rstd,
                          const float* __restrict__ g, const float* __restrict__ b,
                          int M, int F){
    int idx = blockIdx.x * blockDim.x + threadIdx.x;
    if (idx >= M * F) return;
    int f = idx % F;
    Y[idx] = g[f] * (X[idx] - mean[f]) * rstd[f] + b[f];
}

__device__ __forceinline__ float x2col(int row, int c,
        const float* op, const float* ex, const float* last, const float* samp){
    if (c < 15)  return op[(size_t)row * 15 + c];
    if (c < 123) return ex[(size_t)row * 108 + (c - 15)];
    if (c < 379) return last[(size_t)row * 256 + (c - 123)];
    if (c < 635) return samp[(size_t)row * 256 + (c - 379)];
    return 0.f;
}

__global__ void k_pack_x2(const float* __restrict__ op, const float* __restrict__ ex,
                          const float* __restrict__ last, const float* __restrict__ samp,
                          unsigned* __restrict__ x2pk){
    int idx = blockIdx.x * blockDim.x + threadIdx.x;
    if (idx >= BALL * (D2PAD / 2)) return;
    int row = idx / (D2PAD / 2), kp = idx - row * (D2PAD / 2);
    int c = kp * 2;
    x2pk[idx] = pk2(x2col(row, c, op, ex, last, samp),
                    x2col(row, c + 1, op, ex, last, samp));
}

// sigmoid(V[row,:] . Wo + bo) — one wave per row
__global__ void k_head(const float* __restrict__ V, const float* __restrict__ Wo,
                       const float* __restrict__ bo, float* __restrict__ out){
    int gw = (blockIdx.x * blockDim.x + threadIdx.x) >> 5;  // 0..511
    int lane = threadIdx.x & 31;
    float s = 0.f;
    #pragma unroll
    for (int i = 0; i < 8; i++)
        s += V[(size_t)gw * H2 + lane + 32 * i] * Wo[lane + 32 * i];
    #pragma unroll
    for (int off = 16; off > 0; off >>= 1) s += __shfl_xor(s, off, 32);
    if (lane == 0) out[gw] = 1.f / (1.f + expf(-(s + bo[0])));
}

// ================================================================================
extern "C" void kernel_launch(void* const* d_in, const int* in_sizes, int n_in,
                              void* d_out, int out_size, void* d_ws, size_t ws_size,
                              hipStream_t stream)
{
    const float* operators = (const float*)d_in[0];
    const float* extras    = (const float*)d_in[1];
    const float* cond1     = (const float*)d_in[2];
    const float* cond2     = (const float*)d_in[3];
    const float* samples   = (const float*)d_in[4];
    const float* masks     = (const float*)d_in[5];
    const int*   mapping   = (const int*)d_in[6];
    const float* W1ih = (const float*)d_in[7];
    const float* W1hh = (const float*)d_in[8];
    const float* b1ih = (const float*)d_in[9];
    const float* b1hh = (const float*)d_in[10];
    const float* W2ih = (const float*)d_in[11];
    const float* W2hh = (const float*)d_in[12];
    const float* b2ih = (const float*)d_in[13];
    const float* b2hh = (const float*)d_in[14];
    const float* Wsamp = (const float*)d_in[15];
    const float* bsamp = (const float*)d_in[16];
    const float* Wcond = (const float*)d_in[17];
    const float* bcond = (const float*)d_in[18];
    const float* bn1g = (const float*)d_in[19];
    const float* bn1b = (const float*)d_in[20];
    const float* bn2g = (const float*)d_in[21];
    const float* bn2b = (const float*)d_in[22];
    const float* bn3g = (const float*)d_in[23];
    const float* bn3b = (const float*)d_in[24];
    const float* W21 = (const float*)d_in[25];
    const float* b21 = (const float*)d_in[26];
    const float* W22 = (const float*)d_in[27];
    const float* b22 = (const float*)d_in[28];
    const float* W31 = (const float*)d_in[29];
    const float* b31 = (const float*)d_in[30];
    const float* W32 = (const float*)d_in[31];
    const float* b32 = (const float*)d_in[32];
    const float* Wo1 = (const float*)d_in[33];
    const float* bo1 = (const float*)d_in[34];
    const float* Wo2 = (const float*)d_in[35];
    const float* bo2 = (const float*)d_in[36];
    float* out = (float*)d_out;

    char* base = (char*)d_ws;
    size_t off = 0;
    auto alloc = [&](size_t bytes) -> char* {
        char* p = base + off;
        off += (bytes + 255) & ~(size_t)255;
        return p;
    };

    unsigned* w1ih_pk  = (unsigned*)alloc((size_t)512 * 128 * 4);
    unsigned* w1hh_pk  = (unsigned*)alloc((size_t)512 * 64 * 4);
    unsigned* w2ih_pk  = (unsigned*)alloc((size_t)512 * 320 * 4);
    unsigned* w2hh_pk  = (unsigned*)alloc((size_t)512 * 64 * 4);
    unsigned* wsamp_pk = (unsigned*)alloc((size_t)256 * 512 * 4);
    unsigned* wcond_pk = (unsigned*)alloc((size_t)256 * 64 * 4);
    unsigned* w21_pk   = (unsigned*)alloc((size_t)256 * 64 * 4);
    unsigned* w22_pk   = (unsigned*)alloc((size_t)256 * 64 * 4);
    unsigned* w31_pk   = (unsigned*)alloc((size_t)256 * 128 * 4);
    unsigned* w32_pk   = (unsigned*)alloc((size_t)256 * 128 * 4);
    float* samp_out = (float*)alloc((size_t)BALL * 256 * 4);
    float* lo1      = (float*)alloc((size_t)BALL * 128 * 4);
    float* lo2      = (float*)alloc((size_t)BALL * 128 * 4);
    float* lo1f     = (float*)alloc((size_t)BALL * 256 * 4);
    float* lo2f     = (float*)alloc((size_t)BALL * 256 * 4);
    float* avgb     = (float*)alloc((size_t)BALL * 256 * 4);
    float* lastb    = (float*)alloc((size_t)BALL * 256 * 4);
    unsigned* x2pk  = (unsigned*)alloc((size_t)BALL * 320 * 4);
    float* xw2      = (float*)alloc((size_t)BALL * 512 * 4);
    float* hA = (float*)alloc((size_t)NB * 128 * 4);
    float* cA = (float*)alloc((size_t)NB * 128 * 4);
    float* hB = (float*)alloc((size_t)NB * 128 * 4);
    float* cB = (float*)alloc((size_t)NB * 128 * 4);
    float* obn = (float*)alloc((size_t)NB * 128 * 4);
    float* u1  = (float*)alloc((size_t)NB * 256 * 4);
    float* u2  = (float*)alloc((size_t)NB * 256 * 4);
    float* w1b = (float*)alloc((size_t)NB * 256 * 4);
    float* w2b = (float*)alloc((size_t)NB * 256 * 4);
    float* v1  = (float*)alloc((size_t)NB * 256 * 4);
    float* v2  = (float*)alloc((size_t)NB * 256 * 4);
    float* bnm = (float*)alloc(256 * 4);
    float* bnr = (float*)alloc(256 * 4);

    auto packW = [&](const float* W, unsigned* dst, int Nout, int K, int Kpad) {
        int total = Nout * (Kpad / 2);
        k_pack_w<<<dim3((total + 255) / 256), dim3(256), 0, stream>>>(W, dst, Nout, K, Kpad, total);
    };
    packW(W1ih, w1ih_pk, 512, 256, 256);
    packW(W1hh, w1hh_pk, 512, 128, 128);
    packW(W2ih, w2ih_pk, 512, D2DIM, D2PAD);
    packW(W2hh, w2hh_pk, 512, 128, 128);
    packW(Wsamp, wsamp_pk, 256, 1000, 1024);
    packW(Wcond, wcond_pk, 256, 128, 128);
    packW(W21, w21_pk, 256, 128, 128);
    packW(W22, w22_pk, 256, 128, 128);
    packW(W31, w31_pk, 256, 256, 256);
    packW(W32, w32_pk, 256, 256, 256);

    // samp = relu(samples @ Wsamp^T + bsamp) * mask
    k_gemm_f32A<<<dim3(BALL / 64, 2), dim3(256), 0, stream>>>(
        samples, 1000, BALL, 1000, 1024, wsamp_pk, bsamp, masks, samp_out, 256, 1);

    // condition LSTMs
    size_t smem1 = (size_t)(64 * 512 + 64 * 128) * 4 + (size_t)(64 * 64 + 64 * 16) * 4;
    k_lstm1<<<dim3(BALL / 64), dim3(512), smem1, stream>>>(cond1, w1ih_pk, w1hh_pk, b1ih, b1hh, lo1);
    k_lstm1<<<dim3(BALL / 64), dim3(512), smem1, stream>>>(cond2, w1ih_pk, w1hh_pk, b1ih, b1hh, lo2);

    // relu(lo @ Wcond^T + bcond)
    k_gemm_f32A<<<dim3(BALL / 64, 2), dim3(256), 0, stream>>>(
        lo1, 128, BALL, 128, 128, wcond_pk, bcond, nullptr, lo1f, 256, 1);
    k_gemm_f32A<<<dim3(BALL / 64, 2), dim3(256), 0, stream>>>(
        lo2, 128, BALL, 128, 128, wcond_pk, bcond, nullptr, lo2f, 256, 1);

    // last = BN((lo1f+lo2f)/2)
    k_avg<<<dim3((BALL * 256 + 255) / 256), dim3(256), 0, stream>>>(lo1f, lo2f, avgb, BALL * 256);
    k_bnstats<<<dim3(256), dim3(256), 0, stream>>>(avgb, BALL, 256, bnm, bnr);
    k_bnapply<<<dim3((BALL * 256 + 255) / 256), dim3(256), 0, stream>>>(
        avgb, lastb, bnm, bnr, bn1g, bn1b, BALL, 256);

    // x2 = concat(op, extra, last, samp) packed bf16; xw2 = x2 @ W2ih^T + b2ih
    k_pack_x2<<<dim3((BALL * (D2PAD / 2) + 255) / 256), dim3(256), 0, stream>>>(
        operators, extras, lastb, samp_out, x2pk);
    k_gemm_pkA<<<dim3(BALL / 64, 4), dim3(256), 0, stream>>>(
        x2pk, D2PAD / 2, D2PAD, w2ih_pk, b2ih, xw2, 512);

    // bottom-up tree LSTM over levels
    size_t smemT = (size_t)(64 * 512 + 64 * 128) * 4 + (size_t)(64 * 64) * 4;
    k_tree<<<dim3(NB / 64), dim3(512), smemT, stream>>>(
        xw2 + (size_t)(LV - 1) * NB * G1, mapping, w2hh_pk, b2hh, hA, cA, hB, cB, 1);
    float* hp = hB; float* cp = cB; float* ho = hA; float* co = cA;
    for (int lev = LV - 2; lev >= 0; lev--) {
        k_tree<<<dim3(NB / 64), dim3(512), smemT, stream>>>(
            xw2 + (size_t)lev * NB * G1, mapping + (size_t)lev * NB * 2,
            w2hh_pk, b2hh, hp, cp, ho, co, 0);
        float* t;
        t = hp; hp = ho; ho = t;
        t = cp; cp = co; co = t;
    }
    // hp now holds the final hidden state [NB, 128]

    // o = BN2(h)
    k_bnstats<<<dim3(128), dim3(256), 0, stream>>>(hp, NB, 128, bnm, bnr);
    k_bnapply<<<dim3((NB * 128 + 255) / 256), dim3(256), 0, stream>>>(
        hp, obn, bnm, bnr, bn2g, bn2b, NB, 128);

    // heads
    k_gemm_f32A<<<dim3(NB / 64, 2), dim3(256), 0, stream>>>(
        obn, 128, NB, 128, 128, w21_pk, b21, nullptr, u1, 256, 1);
    k_gemm_f32A<<<dim3(NB / 64, 2), dim3(256), 0, stream>>>(
        obn, 128, NB, 128, 128, w22_pk, b22, nullptr, u2, 256, 1);
    k_bnstats<<<dim3(256), dim3(256), 0, stream>>>(u1, NB, 256, bnm, bnr);
    k_bnapply<<<dim3((NB * 256 + 255) / 256), dim3(256), 0, stream>>>(
        u1, w1b, bnm, bnr, bn3g, bn3b, NB, 256);
    k_bnstats<<<dim3(256), dim3(256), 0, stream>>>(u2, NB, 256, bnm, bnr);
    k_bnapply<<<dim3((NB * 256 + 255) / 256), dim3(256), 0, stream>>>(
        u2, w2b, bnm, bnr, bn3g, bn3b, NB, 256);
    k_gemm_f32A<<<dim3(NB / 64, 2), dim3(256), 0, stream>>>(
        w1b, 256, NB, 256, 256, w31_pk, b31, nullptr, v1, 256, 1);
    k_gemm_f32A<<<dim3(NB / 64, 2), dim3(256), 0, stream>>>(
        w2b, 256, NB, 256, 256, w32_pk, b32, nullptr, v2, 256, 1);
    k_head<<<dim3(64), dim3(256), 0, stream>>>(v1, Wo1, bo1, out);
    k_head<<<dim3(64), dim3(256), 0, stream>>>(v2, Wo2, bo2, out + NB);
    (void)in_sizes; (void)n_in; (void)out_size; (void)ws_size;
}
